// Fp8AmpereLinear_1391569404515
// MI455X (gfx1250) — matrix operand order repaired and checked by reading, not compiled, not run on hardware
//
#include <hip/hip_runtime.h>
#include <stdint.h>

#define K_IN   4096
#define N_OUT  11008
#define M_TOT  2048

#define BLK_M  128
#define BLK_N  128
#define BLK_K  32
#define NK     (K_IN / BLK_K)

typedef __attribute__((ext_vector_type(16))) __bf16       v16bf;
typedef __attribute__((ext_vector_type(8)))  float        v8f;
typedef __attribute__((ext_vector_type(8)))  unsigned int v8u;
typedef __attribute__((ext_vector_type(4)))  unsigned int v4u;
typedef __attribute__((ext_vector_type(8)))  int          v8i;
typedef __attribute__((ext_vector_type(4)))  int          v4i;

// LDS byte address of a __shared__ object (flat shared-aperture low 32 bits == LDS offset).
__device__ __forceinline__ unsigned lds_addr(const void* p) {
  return (unsigned)(uintptr_t)p;
}

// Two f32 -> packed bf16x2 (a -> low, b -> high): bias + one v_perm_b32.
__device__ __forceinline__ unsigned cvt_pk_bf16(float a, float b) {
  unsigned ua = __builtin_bit_cast(unsigned, a) + 0x8000u;
  unsigned ub = __builtin_bit_cast(unsigned, b) + 0x8000u;
  return __builtin_amdgcn_perm(ub, ua, 0x07060302u);
}

// Two FP8-E4M3FN bytes (low byte of each dword) -> packed bf16x2 (exact).
__device__ __forceinline__ unsigned decode2_fp8(unsigned b0, unsigned b1) {
  unsigned f0 = ((b0 & 0x80u) << 24) | ((b0 & 0x7Fu) << 20);
  unsigned f1 = ((b1 & 0x80u) << 24) | ((b1 & 0x7Fu) << 20);
  float r0 = __builtin_bit_cast(float, f0) * 0x1.0p+120f;
  float r1 = __builtin_bit_cast(float, f1) * 0x1.0p+120f;
  return __builtin_amdgcn_perm(__builtin_bit_cast(unsigned, r1),
                               __builtin_bit_cast(unsigned, r0), 0x07060302u);
}

__global__ __launch_bounds__(256)
void fp8_marlin_bf16_wmma(const float* __restrict__ x,
                          const int* __restrict__ wbits,
                          const float* __restrict__ scale,
                          const float* __restrict__ bias,
                          unsigned short* __restrict__ out) {
  __shared__ int            Braw[2][BLK_N][BLK_K];   // raw int32 weights (TDM)       32 KB
  __shared__ float          Araw[2][BLK_M][BLK_K];   // raw f32 x (async-to-LDS)      32 KB
  __shared__ unsigned short Bbf [2][BLK_N][BLK_K];   // dequantized bf16 weights      16 KB
  __shared__ unsigned short Abf [2][BLK_M][BLK_K];   // x converted to bf16           16 KB

  const int tid  = threadIdx.x;
  const int lane = tid & 31;
  const int wave = tid >> 5;          // 8 waves
  const int wm   = wave & 1;          // 2 waves along M (64 rows each)
  const int wn   = wave >> 1;         // 4 waves along N (32 cols each)

  const int mBase = blockIdx.y * BLK_M;
  const int nBase = blockIdx.x * BLK_N;

  // staging ownership: thread -> (row = tid/2, half = tid&1) = 16 K-elements
  const int sRow  = tid >> 1;
  const int sHalf = tid & 1;

  const float* aSrc = x + (size_t)(mBase + sRow) * K_IN + sHalf * 16;
  const int*   bTile0 = wbits + (size_t)nBase * K_IN;     // tile origin row, k=0
  const unsigned arDst0 = lds_addr(&Araw[0][sRow][sHalf * 16]);
  const unsigned arDst1 = lds_addr(&Araw[1][sRow][sHalf * 16]);
  const unsigned brLds0 = lds_addr(&Braw[0][0][0]);
  const unsigned brLds1 = lds_addr(&Braw[1][0][0]);

  // ---- TDM: one descriptor moves the whole 128x32 int32 weight tile ----
  auto tdmB = [&](int buf, int k0) {
    if (wave == 0) {
      unsigned long long ga =
          (unsigned long long)(uintptr_t)(bTile0 + k0);   // tile-start byte address
      v4u g0;
      g0[0] = 1u;                                         // count=1 (valid user D#)
      g0[1] = buf ? brLds1 : brLds0;                      // lds_addr
      g0[2] = (unsigned)(ga & 0xFFFFFFFFu);               // global_addr[31:0]
      g0[3] = (unsigned)((ga >> 32) & 0x01FFFFFFu) | 0x80000000u;  // addr[56:32] | type=2
      v8i g1;
      g1[0] = 0x00020000;                 // data_size=2 (4-byte words), no multicast
      g1[1] = (int)(4096u << 16);         // tensor_dim0 = 4096 words (K)
      g1[2] = (int)(11008u << 16);        // tensor_dim1 = 11008 rows (N)
      g1[3] = (int)(32u << 16);           // tile_dim0 = 32 (K words)
      g1[4] = 128;                        // tile_dim1 = 128 (N rows)
      g1[5] = 4096;                       // tensor_dim0_stride = 4096 words
      g1[6] = 0;
      g1[7] = 0;
      v4i z4 = {0, 0, 0, 0};              // 2D tensor: groups 2/3 unused
      v8i z8 = {0, 0, 0, 0, 0, 0, 0, 0};
      __builtin_amdgcn_tensor_load_to_lds(g0, g1, z4, z4, z8, 0);
    }
  };
  // async copy: own 16 f32 of the x tile straight into LDS (ASYNCcnt, self-owned)
  auto stageA_async = [&](int buf, const float* g) {
    unsigned l = buf ? arDst1 : arDst0;
#pragma unroll
    for (int c = 0; c < 4; ++c) {
      asm volatile("global_load_async_to_lds_b128 %0, %1, off"
                   :: "v"(l + c * 16), "v"(g + c * 4) : "memory");
    }
  };
  // dequant own raw-B region (TDM-landed) fp8 -> bf16
  auto decodeB = [&](int buf) {
    const int* rp = &Braw[buf][sRow][sHalf * 16];
    int4 w0 = ((const int4*)rp)[0];
    int4 w1 = ((const int4*)rp)[1];
    int4 w2 = ((const int4*)rp)[2];
    int4 w3 = ((const int4*)rp)[3];
    int wv[16];
    *(int4*)&wv[0] = w0; *(int4*)&wv[4] = w1; *(int4*)&wv[8] = w2; *(int4*)&wv[12] = w3;
    unsigned pk[8];
#pragma unroll
    for (int t = 0; t < 8; ++t)
      pk[t] = decode2_fp8((unsigned)wv[2 * t], (unsigned)wv[2 * t + 1]);
    *(uint4*)&Bbf[buf][sRow][sHalf * 16]     = make_uint4(pk[0], pk[1], pk[2], pk[3]);
    *(uint4*)&Bbf[buf][sRow][sHalf * 16 + 8] = make_uint4(pk[4], pk[5], pk[6], pk[7]);
  };
  // convert own raw-A region (async-landed) f32 -> bf16
  auto convA = [&](int buf) {
    const float* rp = &Araw[buf][sRow][sHalf * 16];
    float4 f0 = ((const float4*)rp)[0];
    float4 f1 = ((const float4*)rp)[1];
    float4 f2 = ((const float4*)rp)[2];
    float4 f3 = ((const float4*)rp)[3];
    unsigned pk[8];
    pk[0] = cvt_pk_bf16(f0.x, f0.y); pk[1] = cvt_pk_bf16(f0.z, f0.w);
    pk[2] = cvt_pk_bf16(f1.x, f1.y); pk[3] = cvt_pk_bf16(f1.z, f1.w);
    pk[4] = cvt_pk_bf16(f2.x, f2.y); pk[5] = cvt_pk_bf16(f2.z, f2.w);
    pk[6] = cvt_pk_bf16(f3.x, f3.y); pk[7] = cvt_pk_bf16(f3.z, f3.w);
    *(uint4*)&Abf[buf][sRow][sHalf * 16]     = make_uint4(pk[0], pk[1], pk[2], pk[3]);
    *(uint4*)&Abf[buf][sRow][sHalf * 16 + 8] = make_uint4(pk[4], pk[5], pk[6], pk[7]);
  };

  v8f acc[4][2];
#pragma unroll
  for (int i = 0; i < 4; ++i)
#pragma unroll
    for (int j = 0; j < 2; ++j)
      acc[i][j] = (v8f){0.f, 0.f, 0.f, 0.f, 0.f, 0.f, 0.f, 0.f};

  const int aRow = wm * 64;
  const int nCol = wn * 32;
  const int lRow = lane & 15;
  const int kA   = (lane < 16) ? 0 : 8;     // 16-bit A 16x32 lane layout
  const int kB   = (lane < 16) ? 0 : 8;     // 16-bit B 32x16 lane layout

  // ---- prologue: raw tile 0 (TDM + async), decode it, launch raw tile 1 ----
  tdmB(0, 0);
  stageA_async(0, aSrc);
  if (wave == 0) __builtin_amdgcn_s_wait_tensorcnt(0);
  asm volatile("s_wait_asynccnt 0" ::: "memory");
  __syncthreads();                       // raw tile 0 visible
  decodeB(0);
  convA(0);
  tdmB(1, BLK_K);
  stageA_async(1, aSrc + BLK_K);
  __syncthreads();                       // bf16 tile 0 visible; raw tile 1 in flight

  for (int kt = 0; kt < NK; ++kt) {
    const int cur = kt & 1;
    const int nxt = cur ^ 1;

    // ---- compute tile kt: pure ds_load fragments + 8 WMMAs ----
    v16bf aF[4];
#pragma unroll
    for (int i = 0; i < 4; ++i) {
      const unsigned short* ap = &Abf[cur][aRow + i * 16 + lRow][0];
      uint4 q0 = *(const uint4*)(ap + kA);        // K = kA .. kA+7
      uint4 q1 = *(const uint4*)(ap + kA + 16);   // K = kA+16 .. kA+23
      v8u u = {q0.x, q0.y, q0.z, q0.w, q1.x, q1.y, q1.z, q1.w};
      aF[i] = __builtin_bit_cast(v16bf, u);
    }
    v16bf bF[2];
#pragma unroll
    for (int j = 0; j < 2; ++j) {
      const unsigned short* bp = &Bbf[cur][nCol + j * 16 + lRow][kB];
      uint4 q0 = *(const uint4*)(bp);
      uint4 q1 = *(const uint4*)(bp + 16);
      v8u u = {q0.x, q0.y, q0.z, q0.w, q1.x, q1.y, q1.z, q1.w};
      bF[j] = __builtin_bit_cast(v16bf, u);
    }
#pragma unroll
    for (int i = 0; i < 4; ++i)
#pragma unroll
      for (int j = 0; j < 2; ++j)
        acc[i][j] = __builtin_amdgcn_wmma_f32_16x16x32_bf16(
            false, aF[i], false, bF[j], (short)0, acc[i][j], false, false);

    // ---- publish raw tile kt+1, dequant it, launch raw tile kt+2 ----
    if (kt + 1 < NK) {
      asm volatile("s_wait_asynccnt 0" ::: "memory");       // own A region landed
      if (wave == 0) __builtin_amdgcn_s_wait_tensorcnt(0);  // TDM B tile landed
      __syncthreads();                                      // raw kt+1 visible
      decodeB(nxt);
      convA(nxt);
      if (kt + 2 < NK) {
        tdmB(cur, (kt + 2) * BLK_K);
        stageA_async(cur, aSrc + (size_t)(kt + 2) * BLK_K);
      }
      __syncthreads();                                      // bf16 kt+1 visible
    }
  }

  // ---- epilogue: out = bf16(acc * scale[n] + bias[n]) ----
#pragma unroll
  for (int j = 0; j < 2; ++j) {
    const int n  = nBase + nCol + j * 16 + lRow;
    const float sc = scale[n];
    const float bi = bias[n];
#pragma unroll
    for (int i = 0; i < 4; ++i) {
      const int mB = mBase + aRow + i * 16 + ((lane < 16) ? 0 : 8);
#pragma unroll
      for (int r = 0; r < 8; r += 2) {
        float v0 = acc[i][j][r]     * sc + bi;
        float v1 = acc[i][j][r + 1] * sc + bi;
        unsigned pr = cvt_pk_bf16(v0, v1);
        out[(size_t)(mB + r)     * N_OUT + n] = (unsigned short)pr;
        out[(size_t)(mB + r + 1) * N_OUT + n] = (unsigned short)(pr >> 16);
      }
    }
  }
}

extern "C" void kernel_launch(void* const* d_in, const int* in_sizes, int n_in,
                              void* d_out, int out_size, void* d_ws, size_t ws_size,
                              hipStream_t stream) {
  const float* x  = (const float*)d_in[0];
  const int*   wb = (const int*)d_in[1];
  const float* sc = (const float*)d_in[2];
  const float* bi = (const float*)d_in[3];
  unsigned short* out = (unsigned short*)d_out;

  dim3 grid(N_OUT / BLK_N, M_TOT / BLK_M);   // (86, 16)
  fp8_marlin_bf16_wmma<<<grid, dim3(256), 0, stream>>>(x, wb, sc, bi, out);
}
